// net_LSTM_7851200217946
// MI455X (gfx1250) — matrix operand order repaired and checked by reading, not compiled, not run on hardware
//
#include <hip/hip_runtime.h>

typedef __attribute__((ext_vector_type(16))) _Float16 v16h;
typedef __attribute__((ext_vector_type(8)))  _Float16 v8h;
typedef __attribute__((ext_vector_type(8)))  float    v8f;

#define LSTM_SEQ   2048
#define LSTM_BATCH 1024
#define LSTM_IN    10
#define LSTM_H     20

// Compiler-only reordering fence. Workgroup == one wave32 and LDS ops from a
// single wave are processed in order (DScnt in-order), so no s_barrier /
// counter flush is required for the LDS round-trip below.
__device__ __forceinline__ void wave_lds_fence() {
  asm volatile("" ::: "memory");
}

// ---- fast activations (v_tanh_f32 on gfx1250) ------------------------------
__device__ __forceinline__ float fast_tanh(float v) {
#if __has_builtin(__builtin_amdgcn_tanhf)
  return __builtin_amdgcn_tanhf(v);
#else
  float e = __builtin_amdgcn_exp2f(v * -2.88539008177793f); // exp(-2v)
  return 2.0f * __builtin_amdgcn_rcpf(1.0f + e) - 1.0f;
#endif
}

__device__ __forceinline__ float fast_sigmoid(float v) {
#if __has_builtin(__builtin_amdgcn_tanhf)
  return __builtin_fmaf(0.5f, __builtin_amdgcn_tanhf(0.5f * v), 0.5f);
#else
  float e = __builtin_amdgcn_exp2f(v * -1.44269504088896f);
  return __builtin_amdgcn_rcpf(1.0f + e);
#endif
}

// ---- fused bidirectional LSTM ----------------------------------------------
// One wave32 per (direction, 16-batch tile). Gate dim padded 80->128 with
// stride-32 gate groups [i|f|g|o]; K padded 20->32 (recurrent) / 10->32
// (input, with a constant-1 at k==10 carrying fused bias b_ih+b_hh).
__global__ __launch_bounds__(32, 1) void bilstm_fused(
    const float* __restrict__ x,
    const float* __restrict__ h0,
    const float* __restrict__ c0,
    const float* __restrict__ wih_f, const float* __restrict__ whh_f,
    const float* __restrict__ bih_f, const float* __restrict__ bhh_f,
    const float* __restrict__ wih_b, const float* __restrict__ whh_b,
    const float* __restrict__ bih_b, const float* __restrict__ bhh_b,
    float* __restrict__ out)
{
  __shared__ __align__(16) _Float16 hbuf[16 * 32];   // [m][hid] staging for C->A

  const int lane = threadIdx.x & 31;
  const int cix  = lane & 15;     // tile column (and A-layout row M)
  const int lh   = lane >> 4;     // lane half

  const int d  = (int)(blockIdx.x >> 6);       // direction
  const int b0 = (int)(blockIdx.x & 63) * 16;  // batch tile base

  const float* wih = d ? wih_b : wih_f;
  const float* whh = d ? whh_b : whh_f;
  const float* bi  = d ? bih_b : bih_f;
  const float* bh  = d ? bhh_b : bhh_f;

  // --- preload B operands (loop-invariant, live in VGPRs) -------------------
  // B operands as packed here: lane holds column N=lane%16; Bhh elems 0..15 hold
  // K = lh*16 .. lh*16+15; Bih elems 0..7 hold K = lh*8 .. lh*8+7.
  v16h Bih[8], Bhh[8];
#pragma unroll
  for (int T = 0; T < 8; ++T) {
    const int col  = T * 16 + cix;       // padded gate column 0..127
    const int q    = col >> 5;           // gate group: 0=i 1=f 2=g 3=o
    const int rcol = col & 31;           // index within gate group
    const bool cv  = rcol < LSTM_H;
    const int g    = q * LSTM_H + rcol;  // row into (80 x K) weight
    v16h vh, vi;
#pragma unroll
    for (int e = 0; e < 16; ++e) { vh[e] = (_Float16)0.0f; vi[e] = (_Float16)0.0f; }
    if (cv) {
      if (lh == 0) {        // K = 0..15
#pragma unroll
        for (int e = 0; e < 16; ++e) vh[e] = (_Float16)whh[g * LSTM_H + e];
#pragma unroll
        for (int e = 0; e < 8; ++e) vi[e] = (_Float16)wih[g * LSTM_IN + e];
      } else {              // w_hh K = 16..19; w_ih K = 8..9 + bias at k==10
#pragma unroll
        for (int e = 0; e < 4; ++e) vh[e] = (_Float16)whh[g * LSTM_H + 16 + e];
#pragma unroll
        for (int e = 0; e < 2; ++e) vi[e] = (_Float16)wih[g * LSTM_IN + 8 + e];
        vi[2] = (_Float16)(bi[g] + bh[g]);   // bias folded at k==10
      }
    }
    Bhh[T] = vh; Bih[T] = vi;
  }

  // --- load initial state in C-layout: value(m=r+8*lh, hid=hp*16+cix) -------
  float hc[2][8], cc[2][8];
#pragma unroll
  for (int hp = 0; hp < 2; ++hp)
#pragma unroll
    for (int r = 0; r < 8; ++r) {
      const int m   = r + 8 * lh;
      const int hid = hp * 16 + cix;
      float hv = 0.0f, cv2 = 0.0f;
      if (hid < LSTM_H) {
        const int idx = (d * LSTM_BATCH + b0 + m) * LSTM_H + hid;
        hv  = h0[idx];
        cv2 = c0[idx];
      }
      hc[hp][r] = hv; cc[hp][r] = cv2;
    }

  const int t0 = d ? (LSTM_SEQ - 1) : 0;
  const ptrdiff_t xinc = (d ? -1 : 1) * (ptrdiff_t)(LSTM_BATCH * LSTM_IN);
  const ptrdiff_t oinc = (d ? -1 : 1) * (ptrdiff_t)(LSTM_BATCH * 2 * LSTM_H);

  // per-lane x base (advanced by xinc each step); lh==0 lanes read x[0..7],
  // lh==1 lanes read x[8..9] (lh*8 folds the +8).
  const float* xb = x + ((size_t)t0 * LSTM_BATCH + (size_t)(b0 + cix)) * LSTM_IN
                      + lh * 8;
  // per-lane out base: all in-loop store offsets are compile-time constants.
  float* op = out + ((size_t)t0 * LSTM_BATCH + (size_t)(b0 + 8 * lh)) * (2 * LSTM_H)
                  + d * LSTM_H + cix;

  // --- raw x prefetch registers (one timestep ahead) ------------------------
  float2 q0, q1, q2, q3;
  q1 = q2 = q3 = make_float2(0.f, 0.f);
  if (lh == 0) {
    q0 = *(const float2*)(xb + 0);
    q1 = *(const float2*)(xb + 2);
    q2 = *(const float2*)(xb + 4);
    q3 = *(const float2*)(xb + 6);
  } else {
    q0 = *(const float2*)(xb + 0);
  }

  for (int s = 0; s < LSTM_SEQ; ++s) {
    // ---- x tile in A layout (f16 16x32) from prefetched registers ----------
    v16h xa;
#pragma unroll
    for (int e = 0; e < 16; ++e) xa[e] = (_Float16)0.0f;
    if (lh == 0) {
      xa[0] = (_Float16)q0.x; xa[1] = (_Float16)q0.y;
      xa[2] = (_Float16)q1.x; xa[3] = (_Float16)q1.y;
      xa[4] = (_Float16)q2.x; xa[5] = (_Float16)q2.y;
      xa[6] = (_Float16)q3.x; xa[7] = (_Float16)q3.y;
    } else {
      xa[0] = (_Float16)q0.x;       // x[8]
      xa[1] = (_Float16)q0.y;       // x[9]
      xa[2] = (_Float16)1.0f;       // k == 10 -> bias row of Bih
    }

    // ---- prefetch next timestep's x row (latency hidden behind this step) ---
    xb += (s + 1 < LSTM_SEQ) ? xinc : 0;   // last iter: harmless re-read
    if (lh == 0) {
      q0 = *(const float2*)(xb + 0);
      q1 = *(const float2*)(xb + 2);
      q2 = *(const float2*)(xb + 4);
      q3 = *(const float2*)(xb + 6);
    } else {
      q0 = *(const float2*)(xb + 0);
    }

    // ---- h: C layout -> A layout via 1KB LDS round trip ----------------------
    // (single wave: HW keeps DS ops in order; only compiler fences needed)
    wave_lds_fence();
#pragma unroll
    for (int hp = 0; hp < 2; ++hp)
#pragma unroll
      for (int r = 0; r < 8; ++r)
        hbuf[(r + 8 * lh) * 32 + hp * 16 + cix] = (_Float16)hc[hp][r];
    wave_lds_fence();
    v16h ha;
    {
      const v8h* hr = (const v8h*)(hbuf + cix * 32 + lh * 16);
      v8h lo = hr[0];   // K = lh*16 .. lh*16+7
      v8h hi = hr[1];   // K = lh*16+8 .. lh*16+15
#pragma unroll
      for (int e = 0; e < 8; ++e) { ha[e] = lo[e]; ha[8 + e] = hi[e]; }
    }

    // ---- gates = x*W_ih^T + bias  (8 independent WMMAs) ----------------------
    v8f gates[8];
    const v8f zc = {0.f, 0.f, 0.f, 0.f, 0.f, 0.f, 0.f, 0.f};
#pragma unroll
    for (int T = 0; T < 8; ++T)
      gates[T] = __builtin_amdgcn_wmma_f32_16x16x32_f16(
          false, xa, false, Bih[T], (short)0, zc, false, false);

    // ---- gates += h*W_hh^T  (each dep is 8 instructions away) ----------------
#pragma unroll
    for (int T = 0; T < 8; ++T)
      gates[T] = __builtin_amdgcn_wmma_f32_16x16x32_f16(
          false, ha, false, Bhh[T], (short)0, gates[T], false, false);

    // ---- LSTM elementwise (purely per-lane thanks to stride-32 padding) ------
#pragma unroll
    for (int hp = 0; hp < 2; ++hp) {
#pragma unroll
      for (int r = 0; r < 8; ++r) {
        const float iv = gates[0 + hp][r];
        const float fv = gates[2 + hp][r];
        const float gv = gates[4 + hp][r];
        const float ov = gates[6 + hp][r];
        const float cn = fast_sigmoid(fv) * cc[hp][r]
                       + fast_sigmoid(iv) * fast_tanh(gv);
        cc[hp][r] = cn;
        hc[hp][r] = fast_sigmoid(ov) * fast_tanh(cn);
      }
    }

    // ---- stream h_t to output: constant immediate offsets off one pointer ----
#pragma unroll
    for (int r = 0; r < 8; ++r)                       // hp==0: hid=cix (<20)
      __builtin_nontemporal_store(hc[0][r], op + r * (2 * LSTM_H));
    if (cix < LSTM_H - 16) {                          // hp==1: hid=16+cix
#pragma unroll
      for (int r = 0; r < 8; ++r)
        __builtin_nontemporal_store(hc[1][r], op + r * (2 * LSTM_H) + 16);
    }
    op += oinc;
  }

  // ---- final h_n / c_n ------------------------------------------------------
  const size_t OUT_HN = (size_t)LSTM_SEQ * LSTM_BATCH * (2 * LSTM_H);
#pragma unroll
  for (int hp = 0; hp < 2; ++hp) {
    const int hid = hp * 16 + cix;
    if (hid < LSTM_H) {
#pragma unroll
      for (int r = 0; r < 8; ++r) {
        const int m = r + 8 * lh;
        const size_t idx = OUT_HN + (size_t)(d * LSTM_BATCH + b0 + m) * LSTM_H + hid;
        out[idx] = hc[hp][r];
        out[idx + (size_t)2 * LSTM_BATCH * LSTM_H] = cc[hp][r];
      }
    }
  }
}

extern "C" void kernel_launch(void* const* d_in, const int* in_sizes, int n_in,
                              void* d_out, int out_size, void* d_ws, size_t ws_size,
                              hipStream_t stream) {
  (void)in_sizes; (void)n_in; (void)out_size; (void)d_ws; (void)ws_size;
  bilstm_fused<<<dim3(128), dim3(32), 0, stream>>>(
      (const float*)d_in[0],  // x
      (const float*)d_in[1],  // h0
      (const float*)d_in[2],  // c0
      (const float*)d_in[3],  // w_ih_f
      (const float*)d_in[4],  // w_hh_f
      (const float*)d_in[5],  // b_ih_f
      (const float*)d_in[6],  // b_hh_f
      (const float*)d_in[7],  // w_ih_b
      (const float*)d_in[8],  // w_hh_b
      (const float*)d_in[9],  // b_ih_b
      (const float*)d_in[10], // b_hh_b
      (float*)d_out);
}
